// RandLANetMultiHead_59579786330607
// MI455X (gfx1250) — matrix operand order repaired and checked by reading, not compiled, not run on hardware
//
#include <hip/hip_runtime.h>
#include <math.h>

// ---------------------------------------------------------------------------
// RandLA-Net multi-head inference for gfx1250 (MI455X).
// 1x1 convs run as f16-input / f32-accumulate WMMA GEMMs (V_WMMA_F32_16X16X32_F16)
// with BN + LeakyReLU fused into the epilogue. Weight tile staged in LDS
// (f16, zero-padded) -> A fragments via ds_load_b128; activation loads are
// clause-batched branchless global_load_b32. No masking in the hot loop:
// out-of-range rows/cols produce garbage that is never stored, K-tail is
// neutralized by A's zero padding.
// ---------------------------------------------------------------------------

typedef __attribute__((ext_vector_type(16))) _Float16 v16h;
typedef __attribute__((ext_vector_type(8)))  _Float16 v8h;
typedef __attribute__((ext_vector_type(8)))  float    v8f;

#define EPSBN 1e-5f
#define NEGSL 0.2f
#define WAVES_PER_BLK 4     // 128 threads; each wave owns a 16x32 output tile

// ---------------------------------------------------------------------------
// Fused conv1x1 + (optional) BN + (optional) LeakyReLU GEMM.
// X: [bs, CinTot, Ncols]  (read channels [ciOff, ciOff+Cin))
// Y: [bs, CoutTot, Ncols] (write channels [coOff, coOff+Cout))
// W: [Cout, Cin] row-major, Bb: [Cout]; Gm/Bt/Mn/Vr: BN params or nullptr.
// Dynamic LDS: 16 * CinP * sizeof(_Float16), CinP = round-up-32(Cin).
// ---------------------------------------------------------------------------
__global__ __launch_bounds__(128)
void k_convbn(const float* __restrict__ X, float* __restrict__ Y,
              const float* __restrict__ W, const float* __restrict__ Bb,
              const float* __restrict__ Gm, const float* __restrict__ Bt,
              const float* __restrict__ Mn, const float* __restrict__ Vr,
              int Cin, int CinTot, int ciOff,
              int Cout, int CoutTot, int coOff,
              int Ncols, int act)
{
  extern __shared__ _Float16 smem[];        // [16][CinP] zero-padded f16 weights
  const int CinP = (Cin + 31) & ~31;

  const int lane = threadIdx.x & 31;
  const int wave = threadIdx.x >> 5;
  const int b    = blockIdx.z;
  const int col0 = (blockIdx.x * WAVES_PER_BLK + wave) * 32;  // two 16-col tiles
  const int row0 = blockIdx.y * 16;
  const int m    = lane & 15;
  const int half = lane >> 4;

  // ---- cooperative weight-tile stage (one-time; branchy is fine here) ----
  for (int r = 0; r < 16; ++r) {
    const int row = row0 + r;
    for (int k = threadIdx.x; k < CinP; k += 128) {
      const float w = (row < Cout && k < Cin) ? W[(size_t)row * Cin + k] : 0.0f;
      smem[r * CinP + k] = (_Float16)w;
    }
  }
  __syncthreads();

  const float* __restrict__ Xb = X + (size_t)b * CinTot * Ncols + (size_t)ciOff * Ncols;
  float* __restrict__ Yb = Y + (size_t)b * CoutTot * Ncols + (size_t)coOff * Ncols;

  // Clamped per-lane column bases: loads always in-bounds; out-of-range
  // columns compute garbage that is never stored.
  const int  colB0 = col0 + m;
  const int  colB1 = col0 + 16 + m;
  const bool c0Ok  = colB0 < Ncols;
  const bool c1Ok  = colB1 < Ncols;
  const float* __restrict__ bP0 = Xb + (c0Ok ? colB0 : (Ncols - 1));
  const float* __restrict__ bP1 = Xb + (c1Ok ? colB1 : (Ncols - 1));

  const _Float16* __restrict__ aRow = smem + m * CinP + half * 8;

  v8f acc0 = {}, acc1 = {};
  const int kFull = Cin & ~31;              // full 32-slabs (no clamping needed)

  for (int k0 = 0; k0 < kFull; k0 += 32) {
    if ((k0 + 32) < Cin) {
      __builtin_prefetch(bP0 + (size_t)(k0 + 32) * Ncols, 0, 1);
    }
    // A fragment: two ds_load_b128 from the zero-padded LDS tile
    const v8h lo = *(const v8h*)(aRow + k0);
    const v8h hi = *(const v8h*)(aRow + k0 + 16);
    v16h av;
#pragma unroll
    for (int e = 0; e < 8; ++e) { av[e] = lo[e]; av[e + 8] = hi[e]; }

    v16h bv0, bv1;
#pragma unroll
    for (int e = 0; e < 16; ++e) {
      const size_t off = (size_t)(k0 + half * 8 + (e & 7) + ((e >> 3) << 4)) * Ncols;
      bv0[e] = (_Float16)bP0[off];
      bv1[e] = (_Float16)bP1[off];
    }
    acc0 = __builtin_amdgcn_wmma_f32_16x16x32_f16(false, av, false, bv0,
                                                  (short)0, acc0, false, false);
    acc1 = __builtin_amdgcn_wmma_f32_16x16x32_f16(false, av, false, bv1,
                                                  (short)0, acc1, false, false);
  }

  if (kFull < Cin) {                        // single tail slab, clamped B rows;
    const int k0 = kFull;                   // A is zero in k>=Cin -> no pollution
    const v8h lo = *(const v8h*)(aRow + k0);
    const v8h hi = *(const v8h*)(aRow + k0 + 16);
    v16h av;
#pragma unroll
    for (int e = 0; e < 8; ++e) { av[e] = lo[e]; av[e + 8] = hi[e]; }

    v16h bv0, bv1;
#pragma unroll
    for (int e = 0; e < 16; ++e) {
      const int kb  = k0 + half * 8 + (e & 7) + ((e >> 3) << 4);
      const int kbC = (kb < Cin) ? kb : (Cin - 1);
      const size_t off = (size_t)kbC * Ncols;
      bv0[e] = (_Float16)bP0[off];
      bv1[e] = (_Float16)bP1[off];
    }
    acc0 = __builtin_amdgcn_wmma_f32_16x16x32_f16(false, av, false, bv0,
                                                  (short)0, acc0, false, false);
    acc1 = __builtin_amdgcn_wmma_f32_16x16x32_f16(false, av, false, bv1,
                                                  (short)0, acc1, false, false);
  }

  // Epilogue: y = acc*scale + (bias*scale + beta - mean*scale); optional lrelu.
#pragma unroll
  for (int r = 0; r < 8; ++r) {
    const int co = row0 + r + 8 * half;     // D layout: VGPR r -> row r + 8*half
    if (co < Cout) {
      float scale = 1.0f, shift = 0.0f;
      if (Gm) {
        const float s = Gm[co] * rsqrtf(Vr[co] + EPSBN);
        scale = s;
        shift = Bt[co] - Mn[co] * s;
      }
      const float bias = Bb ? Bb[co] : 0.0f;
      float v0 = acc0[r] * scale + bias * scale + shift;
      float v1 = acc1[r] * scale + bias * scale + shift;
      if (act) {
        v0 = (v0 >= 0.0f) ? v0 : NEGSL * v0;
        v1 = (v1 >= 0.0f) ? v1 : NEGSL * v1;
      }
      if (c0Ok) Yb[(size_t)co * Ncols + colB0] = v0;
      if (c1Ok) Yb[(size_t)co * Ncols + colB1] = v1;
    }
  }
}

// ---------------------------------------------------------------------------
// relative_pos_encoding: xyz [bs,3,N], idx [bs,N,K] -> out [bs,10,N,K]
// channels: 0 dist, 1-3 rel, 4-6 tile, 7-9 neighbor
// ---------------------------------------------------------------------------
__global__ __launch_bounds__(256)
void k_relpos(const float* __restrict__ xyz, const int* __restrict__ idx,
              float* __restrict__ out, int N, int K)
{
  const int t = blockIdx.x * blockDim.x + threadIdx.x;  // over N*K
  const int b = blockIdx.z;
  if (t >= N * K) return;
  const int n = t / K;
  const float* xb = xyz + (size_t)b * 3 * N;
  const int j = idx[(size_t)b * N * K + t];
  const float tx = xb[n],     ty = xb[N + n],  tz = xb[2 * N + n];
  const float nx = xb[j],     ny = xb[N + j],  nz = xb[2 * N + j];
  const float rx = tx - nx,   ry = ty - ny,    rz = tz - nz;
  const float d  = sqrtf(rx * rx + ry * ry + rz * rz);
  float* ob = out + (size_t)b * 10 * N * K;
  const size_t s = (size_t)N * K;
  ob[t]         = d;
  ob[s + t]     = rx;  ob[2 * s + t] = ry;  ob[3 * s + t] = rz;
  ob[4 * s + t] = tx;  ob[5 * s + t] = ty;  ob[6 * s + t] = tz;
  ob[7 * s + t] = nx;  ob[8 * s + t] = ny;  ob[9 * s + t] = nz;
}

// gather_neighbour into channel window of a wider buffer (fused concat)
// f [bs,C,Nf], idx [bs,M,K] -> out [bs,CoutTot,M,K] at channel coOff
__global__ __launch_bounds__(256)
void k_gather(const float* __restrict__ f, const int* __restrict__ idx,
              float* __restrict__ out, int C, int Nf, int M, int K,
              int CoutTot, int coOff)
{
  const int t = blockIdx.x * blockDim.x + threadIdx.x;  // over M*K
  const int b = blockIdx.z;
  if (t >= M * K) return;
  const int j = idx[(size_t)b * M * K + t];
  const float* fb = f + (size_t)b * C * Nf;
  float* ob = out + (size_t)b * CoutTot * M * K + (size_t)coOff * M * K;
  for (int c = 0; c < C; ++c)
    ob[(size_t)c * M * K + t] = fb[(size_t)c * Nf + j];
}

// att_pool core: per (b,c,n) softmax over K of scores, weighted-sum of feat
__global__ __launch_bounds__(256)
void k_softmax_agg(const float* __restrict__ feat, const float* __restrict__ sc,
                   float* __restrict__ out, int C, int N, int K)
{
  const int t = blockIdx.x * blockDim.x + threadIdx.x;  // over C*N
  const int b = blockIdx.z;
  if (t >= C * N) return;
  const float* sb = sc   + (size_t)b * C * N * K + (size_t)t * K;
  const float* fb = feat + (size_t)b * C * N * K + (size_t)t * K;
  float mx = -INFINITY;
  for (int k = 0; k < K; ++k) mx = fmaxf(mx, sb[k]);
  float den = 0.0f, num = 0.0f;
  for (int k = 0; k < K; ++k) {
    const float e = __expf(sb[k] - mx);
    den += e;
    num += e * fb[k];
  }
  out[(size_t)b * C * N + t] = num / den;
}

// max over gathered neighbors: out[b,c,m] = max_k f[b,c,idx[b,m,k]]
__global__ __launch_bounds__(256)
void k_maxdown(const float* __restrict__ f, const int* __restrict__ idx,
               float* __restrict__ out, int C, int Nf, int M, int K)
{
  const int t = blockIdx.x * blockDim.x + threadIdx.x;  // over C*M
  const int b = blockIdx.z;
  if (t >= C * M) return;
  const int c = t / M, m = t % M;
  const int* ib = idx + (size_t)b * M * K + (size_t)m * K;
  const float* fb = f + (size_t)b * C * Nf + (size_t)c * Nf;
  float v = -INFINITY;
  for (int k = 0; k < K; ++k) v = fmaxf(v, fb[ib[k]]);
  out[(size_t)b * C * M + t] = v;
}

// nearest upsample into channel window (fused concat): idx [bs,M,1]
__global__ __launch_bounds__(256)
void k_upsample(const float* __restrict__ f, const int* __restrict__ idx,
                float* __restrict__ out, int C, int Nf, int M,
                int CoutTot, int coOff)
{
  const int t = blockIdx.x * blockDim.x + threadIdx.x;  // over C*M
  const int b = blockIdx.z;
  if (t >= C * M) return;
  const int c = t / M, m = t % M;
  const int j = idx[(size_t)b * M + m];
  out[(size_t)b * CoutTot * M + (size_t)(coOff + c) * M + m] =
      f[(size_t)b * C * Nf + (size_t)c * Nf + j];
}

// channel-offset copy (skip branch of decoder concat)
__global__ __launch_bounds__(256)
void k_copych(const float* __restrict__ src, float* __restrict__ dst,
              int C, int N, int CoutTot, int coOff)
{
  const int t = blockIdx.x * blockDim.x + threadIdx.x;  // over C*N
  const int b = blockIdx.z;
  if (t >= C * N) return;
  dst[(size_t)b * CoutTot * N + (size_t)coOff * N + t] = src[(size_t)b * C * N + t];
}

// out = lrelu(a + c), flat
__global__ __launch_bounds__(256)
void k_add_lrelu(const float* __restrict__ a, const float* __restrict__ c,
                 float* __restrict__ o, long n)
{
  const long i = (long)blockIdx.x * blockDim.x + threadIdx.x;
  if (i >= n) return;
  const float v = a[i] + c[i];
  o[i] = (v >= 0.0f) ? v : NEGSL * v;
}

// ---------------------------------------------------------------------------
// Host orchestration
// ---------------------------------------------------------------------------
struct ConvP { const float *w, *b, *g, *bt, *mn, *vr; };
struct AttP  { ConvP fc; ConvP mlp; };

extern "C" void kernel_launch(void* const* d_in, const int* in_sizes, int n_in,
                              void* d_out, int out_size, void* d_ws, size_t ws_size,
                              hipStream_t stream)
{
  (void)in_sizes; (void)n_in; (void)out_size; (void)ws_size;
  const int BS = 8, K = 16;
  static const int Ns[5] = {16384, 4096, 1024, 256, 64};
  static const int SN[5] = {4096, 1024, 256, 64, 32};
  static const int DO[5] = {16, 64, 128, 256, 512};

  // ---- data inputs (setup_inputs dict order) ----
  const float* features = (const float*)d_in[0];
  const float* xyz[5]; const int* neigh[5]; const int* sub[5]; const int* interp[5];
  for (int i = 0; i < 5; ++i) {
    xyz[i]    = (const float*)d_in[1 + 4 * i];
    neigh[i]  = (const int*)  d_in[2 + 4 * i];
    sub[i]    = (const int*)  d_in[3 + 4 * i];
    interp[i] = (const int*)  d_in[4 + 4 * i];
  }

  // ---- params: jax pytree (alphabetical) flatten, leaves start at index 21 ----
  // conv leaves: b, bn.beta, bn.gamma, bn.mean, bn.var, w
  int cur = 21;
  auto F = [&](int i) { return (const float*)d_in[i]; };
  auto takeConv = [&]() {
    ConvP q;
    q.b = F(cur); q.bt = F(cur + 1); q.g = F(cur + 2);
    q.mn = F(cur + 3); q.vr = F(cur + 4); q.w = F(cur + 5);
    cur += 6; return q;
  };
  auto takeAtt = [&]() {
    AttP a;
    a.fc.b = F(cur); a.fc.w = F(cur + 1);
    a.fc.g = a.fc.bt = a.fc.mn = a.fc.vr = nullptr;
    cur += 2;
    a.mlp = takeConv();
    return a;
  };

  // top-level keys sorted: dec, dec0, enc, heads, stem
  ConvP decP[5];
  for (int j = 0; j < 5; ++j) decP[j] = takeConv();
  ConvP dec0P = takeConv();
  struct Stage { AttP att1, att2; ConvP lmlp1, lmlp2, mlp1, mlp2, sc; } st[5];
  for (int i = 0; i < 5; ++i) {
    // enc stage keys sorted: lfa (att1, att2, mlp1, mlp2), mlp1, mlp2, shortcut
    st[i].att1 = takeAtt(); st[i].att2 = takeAtt();
    st[i].lmlp1 = takeConv(); st[i].lmlp2 = takeConv();
    st[i].mlp1 = takeConv(); st[i].mlp2 = takeConv(); st[i].sc = takeConv();
  }
  struct HeadP { ConvP c1, c2, out; } hd[5];
  for (int j = 0; j < 5; ++j) {
    hd[j].c1 = takeConv(); hd[j].c2 = takeConv();
    hd[j].out.b = F(cur); hd[j].out.w = F(cur + 1);
    hd[j].out.g = hd[j].out.bt = hd[j].out.mn = hd[j].out.vr = nullptr;
    cur += 2;
  }
  ConvP stemP = takeConv();

  // ---- workspace arena (floats); ~460 MB total ----
  float* ws = (float*)d_ws;
  size_t o = 0;
  auto alloc = [&](size_t n) { float* p = ws + o; o += n; return p; };
  float* B1 = alloc((size_t)10 * 16384 * 16 * BS);   // relpos / decoder scratch
  float* B2 = alloc((size_t)33554432);               // d_out @ (N,K) ping
  float* B3 = alloc((size_t)33554432);               // d_out @ (N,K) pong
  float* S1 = alloc((size_t)1048576);                // d/2  @ N
  float* S2 = alloc((size_t)2097152);                // d    @ N (pooled pre-mlp)
  float* S3 = alloc((size_t)1048576);                // d/2  @ N (agg)
  float* S4 = alloc((size_t)2097152);                // d    @ N (lfa out)
  float* S5 = alloc((size_t)4194304);                // 2d   @ N
  float* S6 = alloc((size_t)4194304);                // 2d   @ N (shortcut)
  float* stemB = alloc((size_t)8 * 16384 * BS);
  float* F0 = alloc((size_t)4194304);                // enc[0] skip, 32@16384
  float* P[5];
  for (int i = 0; i < 5; ++i) P[i] = alloc((size_t)2 * DO[i] * SN[i] * BS);
  float* Tout = alloc((size_t)4194304);              // pre-downsample stage output

  // ---- launch helpers ----
  auto convbn = [&](const float* X, float* Y, const ConvP& p,
                    int Cin, int CinTot, int ciOff,
                    int Cout, int CoutTot, int coOff, int Ncols, int act) {
    dim3 g((Ncols + 32 * WAVES_PER_BLK - 1) / (32 * WAVES_PER_BLK),
           (Cout + 15) / 16, BS);
    const int CinP = (Cin + 31) & ~31;
    const size_t shbytes = (size_t)16 * CinP * sizeof(_Float16);
    k_convbn<<<g, 128, shbytes, stream>>>(X, Y, p.w, p.b, p.g, p.bt, p.mn, p.vr,
                                          Cin, CinTot, ciOff, Cout, CoutTot, coOff,
                                          Ncols, act);
  };
  auto ew = [&](int work) { return dim3((work + 255) / 256, 1, BS); };

  // ======================= stem =======================
  convbn(features, stemB, stemP, 6, 6, 0, 8, 8, 0, Ns[0], 1);

  // ======================= encoder =======================
  const float* featIn = stemB;
  int din = 8;
  for (int i = 0; i < 5; ++i) {
    const int N = Ns[i], M = SN[i], dout = DO[i], d2 = dout / 2, NK = N * K;
    const Stage& S = st[i];

    // f = mlp1(feature)
    convbn(featIn, S1, S.mlp1, din, din, 0, d2, d2, 0, N, 1);
    // f_xyz = relpos
    k_relpos<<<ew(NK), 256, 0, stream>>>(xyz[i], neigh[i], B1, N, K);
    // f_xyz1 = lfa.mlp1(f_xyz) -> B2 channels [d2, dout)
    convbn(B1, B2, S.lmlp1, 10, 10, 0, d2, dout, d2, NK, 1);
    // gather(f) -> B2 channels [0, d2)   (fused concat)
    k_gather<<<ew(NK), 256, 0, stream>>>(S1, neigh[i], B2, d2, N, N, K, dout, 0);
    // att1: scores = fc(fc_tensor); pool; mlp
    convbn(B2, B3, S.att1.fc, dout, dout, 0, dout, dout, 0, NK, 0);
    k_softmax_agg<<<ew(dout * N), 256, 0, stream>>>(B2, B3, S2, dout, N, K);
    convbn(S2, S3, S.att1.mlp, dout, dout, 0, d2, d2, 0, N, 1);
    // f_xyz2 = lfa.mlp2(f_xyz1)  (read B2[d2..], write B3[d2..])
    convbn(B2, B3, S.lmlp2, d2, dout, d2, d2, dout, d2, NK, 1);
    // gather(agg) -> B3 channels [0, d2)
    k_gather<<<ew(NK), 256, 0, stream>>>(S3, neigh[i], B3, d2, N, N, K, dout, 0);
    // att2
    convbn(B3, B2, S.att2.fc, dout, dout, 0, dout, dout, 0, NK, 0);
    k_softmax_agg<<<ew(dout * N), 256, 0, stream>>>(B3, B2, S2, dout, N, K);
    convbn(S2, S4, S.att2.mlp, dout, dout, 0, dout, dout, 0, N, 1);
    // f = mlp2(lfa_out) (no act); sc = shortcut(feature) (no act); lrelu(f+sc)
    convbn(S4, S5, S.mlp2, dout, dout, 0, 2 * dout, 2 * dout, 0, N, 0);
    convbn(featIn, S6, S.sc, din, din, 0, 2 * dout, 2 * dout, 0, N, 0);
    float* so = (i == 0) ? F0 : Tout;
    {
      const long tot = (long)BS * 2 * dout * N;
      k_add_lrelu<<<dim3((unsigned)((tot + 255) / 256)), 256, 0, stream>>>(S5, S6, so, tot);
    }
    // max downsample -> persistent skip / next-stage input
    k_maxdown<<<ew(2 * dout * M), 256, 0, stream>>>(so, sub[i], P[i], 2 * dout, N, M, K);
    featIn = P[i];
    din = 2 * dout;
  }

  // ======================= decoder =======================
  float* featDA = B2;
  float* featDB = B2 + (size_t)8388608;
  float* catB   = B3;                        // concat input (<= 8.4M floats)
  float* H1     = B3 + (size_t)8388608;      // head c1 out
  float* H2     = B3 + (size_t)16777216;     // head c2 out

  convbn(P[4], featDA, dec0P, 1024, 1024, 0, 1024, 1024, 0, SN[4], 1);

  const float* skipSrc[5] = {P[3], P[2], P[1], P[0], F0};
  const int cSkip[5]  = {512, 256, 128, 32, 32};
  const int NL[5]     = {64, 256, 1024, 4096, 16384};
  const int dcOut[5]  = {512, 256, 128, 32, 32};

  const float* fd = featDA;
  int cPrev = 1024, nfPrev = SN[4];
  float* out = (float*)d_out;
  size_t outOff = 0;

  for (int j = 0; j < 5; ++j) {
    const int Nl = NL[j];
    const int ctot = cSkip[j] + cPrev;
    // concat(skip, upsample(feat))
    k_copych<<<ew(cSkip[j] * Nl), 256, 0, stream>>>(skipSrc[j], catB, cSkip[j], Nl, ctot, 0);
    k_upsample<<<ew(cPrev * Nl), 256, 0, stream>>>(fd, interp[4 - j], catB,
                                                   cPrev, nfPrev, Nl, ctot, cSkip[j]);
    float* fo = (fd == featDA) ? featDB : featDA;
    convbn(catB, fo, decP[j], ctot, ctot, 0, dcOut[j], dcOut[j], 0, Nl, 1);
    // head: c1 -> c2 -> out (dropout is identity at inference)
    convbn(fo, H1, hd[j].c1, dcOut[j], dcOut[j], 0, 64, 64, 0, Nl, 1);
    convbn(H1, H2, hd[j].c2, 64, 64, 0, 32, 32, 0, Nl, 1);
    convbn(H2, out + outOff, hd[j].out, 32, 32, 0, 13, 13, 0, Nl, 0);
    outOff += (size_t)BS * 13 * Nl;
    fd = fo; cPrev = dcOut[j]; nfPrev = Nl;
  }
}
